// STM_35201551958058
// MI455X (gfx1250) — matrix operand order repaired and checked by reading, not compiled, not run on hardware
//
#include <hip/hip_runtime.h>
#include <math.h>

#define B_ 2
#define T_ 24
#define N_ 1500
#define E_ 2000
#define M_ 4000
#define G_ 48            // B_*T_
#define D_ 64
#define MT_ 4            // M-subtiles (16 rows each) per wave => 64-row macro tile

typedef __attribute__((ext_vector_type(16))) __bf16 v16bf;
typedef __attribute__((ext_vector_type(8)))  float  v8f;

// ---------------------------------------------------------------- helpers
static __device__ inline void atomicMaxF(float* addr, float val) {
  int* ai = (int*)addr;
  int cur = __float_as_int(*addr);
  while (__int_as_float(cur) < val) {
    int prev = atomicCAS(ai, cur, __float_as_int(val));
    if (prev == cur) break;
    cur = prev;
  }
}

static __device__ inline v16bf cvt16(float4 p0, float4 p1, float4 p2, float4 p3) {
  v16bf a;
  a[0]=(__bf16)p0.x;  a[1]=(__bf16)p0.y;  a[2]=(__bf16)p0.z;  a[3]=(__bf16)p0.w;
  a[4]=(__bf16)p1.x;  a[5]=(__bf16)p1.y;  a[6]=(__bf16)p1.z;  a[7]=(__bf16)p1.w;
  a[8]=(__bf16)p2.x;  a[9]=(__bf16)p2.y;  a[10]=(__bf16)p2.z; a[11]=(__bf16)p2.w;
  a[12]=(__bf16)p3.x; a[13]=(__bf16)p3.y; a[14]=(__bf16)p3.z; a[15]=(__bf16)p3.w;
  return a;
}

// ---------------------------------------------------------------- WMMA GEMM
// C[g, i, j] = act( sum_k A[g?, i, k] * B[k, j] + bias[j] )
// A row-major [rows,K] (batch stride sA, 0 => shared).
// B is supplied TRANSPOSED: Bt[j, k], row length K (batch stride sB) so each
// lane's 16-deep B fragment is 4 contiguous b128 loads.
// C: transC==0 -> Cg[i*ldC + j]; transC==1 -> Cg[j*ldC + i]  (batch stride sC).
// One wave computes a 64x16 C macro-tile: all 4 A fragments are materialized
// before the 4 back-to-back v_wmma_f32_16x16x32_bf16 (distinct regs, no NOPs).
__global__ void __launch_bounds__(32)
gemm_bf16_wmma(const float* __restrict__ A, long sA,
               const float* __restrict__ Bt, long sB,
               const float* __restrict__ bias,
               float* __restrict__ C, long sC, int ldC, int transC,
               int rows, int K, int J, int relu)
{
  const int ti = blockIdx.x;              // 64-row macro tile index
  const int tj = blockIdx.y, g = blockIdx.z;
  const float* Ag = A + (long)g * sA;
  const float* Bg = Bt + (long)g * sB;
  float*       Cg = C + (long)g * sC;

  const int lane = threadIdx.x;
  const int half = lane >> 4;             // 0: lanes 0-15, 1: lanes 16-31
  const int l16  = lane & 15;
  const int col  = tj * 16 + l16;         // B/C column this lane holds
  const int ak0  = half * 16;             // A K-base (contiguous k = 16*half + e)
  const int bk0  = half * 16;             // B K-base (contiguous k = 16*half + e)

  const float* Arow[MT_];
#pragma unroll
  for (int m = 0; m < MT_; ++m) {
    int r = ti * (16 * MT_) + m * 16 + l16;
    int rc = r < rows ? r : 0;            // clamp => loads always in-bounds
    Arow[m] = Ag + (long)rc * K + ak0;
  }
  const float* BtCol = Bg + (long)col * K + bk0;

  v8f acc[MT_];
#pragma unroll
  for (int m = 0; m < MT_; ++m) acc[m] = (v8f){0.f,0.f,0.f,0.f,0.f,0.f,0.f,0.f};

  const int Kfull = K & ~31;
  int kk = 0;
  for (; kk < Kfull; kk += 32) {          // ---- guard-free main loop
    const float4* b4 = (const float4*)(BtCol + kk);
    v16bf b = cvt16(b4[0], b4[1], b4[2], b4[3]);
    v16bf af[MT_];
#pragma unroll
    for (int m = 0; m < MT_; ++m) {
      const float4* a4 = (const float4*)(Arow[m] + kk);
      af[m] = cvt16(a4[0], a4[1], a4[2], a4[3]);  // K = ak0+0..15
    }
#pragma unroll
    for (int m = 0; m < MT_; ++m)
      acc[m] = __builtin_amdgcn_wmma_f32_16x16x32_bf16(false, af[m], false, b,
                                                       (short)0, acc[m], false, false);
  }
  if (kk < K) {                           // ---- single guarded tail step
    v16bf b;
#pragma unroll
    for (int e = 0; e < 16; ++e) {
      int k = kk + bk0 + e;
      b[e] = (__bf16)(k < K ? BtCol[kk + e] : 0.f);
    }
    v16bf af[MT_];
#pragma unroll
    for (int m = 0; m < MT_; ++m) {
#pragma unroll
      for (int e = 0; e < 16; ++e) {
        int k1 = kk + ak0 + e;
        af[m][e]     = (__bf16)(k1 < K ? Arow[m][kk + e] : 0.f);
      }
    }
#pragma unroll
    for (int m = 0; m < MT_; ++m)
      acc[m] = __builtin_amdgcn_wmma_f32_16x16x32_bf16(false, af[m], false, b,
                                                       (short)0, acc[m], false, false);
  }
  // ---- epilogue: C M = v + half*8 (+16*m), N = l16
  float bcol = bias ? bias[col] : 0.f;
#pragma unroll
  for (int m = 0; m < MT_; ++m) {
#pragma unroll
    for (int v = 0; v < 8; ++v) {
      int r = ti * (16 * MT_) + m * 16 + half * 8 + v;
      if (r < rows) {
        float val = acc[m][v] + bcol;
        if (relu) val = val > 0.f ? val : 0.f;
        long off = transC ? ((long)col * ldC + r) : ((long)r * ldC + col);
        Cg[off] = val;
      }
    }
  }
}

// ---------------------------------------------------------------- misc kernels
__global__ void fill_k(float* p, float v, long n) {
  long i = (long)blockIdx.x * blockDim.x + threadIdx.x;
  if (i < n) p[i] = v;
}

// dst[j*K + k] = src[k*J + j]  (weight transpose for the GEMM B operand)
__global__ void transpose_k(const float* __restrict__ src, float* __restrict__ dst,
                            int K, int J)
{
  long idx = (long)blockIdx.x * blockDim.x + threadIdx.x;
  if (idx >= (long)K * J) return;
  int j = (int)(idx % J); long k = idx / J;
  dst[(long)j * K + k] = src[idx];
}

// generic tiny encoder: dst = relu(src @ W + b); captures pre-relu t==T-1 row
__global__ void encode_k(const float* __restrict__ src, const float* __restrict__ W,
                         const float* __restrict__ bvec, float* __restrict__ dst,
                         float* __restrict__ res, int R, int din, int dout)
{
  long idx = (long)blockIdx.x * blockDim.x + threadIdx.x;
  long total = (long)G_ * R * dout;
  if (idx >= total) return;
  int d = (int)(idx % dout);
  long gi = idx / dout;
  const float* row = src + gi * din;
  float v = bvec[d];
  for (int c = 0; c < din; ++c) v += row[c] * W[c * dout + d];
  long g = gi / R;
  long i = gi % R;
  int  t = (int)(g % T_);
  long b = g / T_;
  if (res && t == T_ - 1) res[(b * R + i) * (long)dout + d] = v;
  dst[idx] = v > 0.f ? v : 0.f;
}

// copy src[G*rows, cols] into dst rows of width ldD at column colOff
__global__ void pack_k(const float* __restrict__ src, float* __restrict__ dst,
                       long rows, int cols, int ldD, int colOff)
{
  long idx = (long)blockIdx.x * blockDim.x + threadIdx.x;
  long total = rows * cols;
  if (idx >= total) return;
  long r = idx / cols; int c = (int)(idx % cols);
  dst[r * ldD + colOff + c] = src[idx];
}

__global__ void wmask_ne_k(const float* __restrict__ w, const float* __restrict__ p,
                           const float* __restrict__ inci, float* __restrict__ out)
{
  long idx = (long)blockIdx.x * blockDim.x + threadIdx.x;
  if (idx >= (long)N_ * E_) return;
  out[idx] = w[idx] * fabsf(inci[idx]) + p[idx];
}

__global__ void wmask_en_k(const float* __restrict__ w, const float* __restrict__ p,
                           const float* __restrict__ inci, float* __restrict__ out)
{
  long idx = (long)blockIdx.x * blockDim.x + threadIdx.x;
  if (idx >= (long)E_ * N_) return;
  long e = idx / N_, n = idx % N_;
  out[idx] = w[idx] * fabsf(inci[n * (long)E_ + e]) + p[idx];
}

// ---------------------------------------------------------------- GAT kernels
__global__ void gat_dots_k(const float* __restrict__ Z, const float* __restrict__ as_,
                           const float* __restrict__ ad_, float* __restrict__ es,
                           float* __restrict__ ed, int n)
{
  long idx = (long)blockIdx.x * blockDim.x + threadIdx.x;
  if (idx >= (long)G_ * n) return;
  const float* z = Z + idx * D_;
  float s = 0.f, d2 = 0.f;
#pragma unroll 8
  for (int d = 0; d < D_; ++d) { s += z[d] * as_[d]; d2 += z[d] * ad_[d]; }
  es[idx] = s; ed[idx] = d2;
}

__global__ void gat_alpha_k(const float* __restrict__ es, const float* __restrict__ ed,
                            const int* __restrict__ ind, int Elist, int Etot, int n,
                            float* __restrict__ alpha, float* __restrict__ mbuf)
{
  long idx = (long)blockIdx.x * blockDim.x + threadIdx.x;
  if (idx >= (long)G_ * Etot) return;
  long g = idx / Etot; int j = (int)(idx % Etot);
  int s = j < Elist ? ind[j] : j - Elist;
  int d = j < Elist ? ind[Elist + j] : j - Elist;
  float a = es[g * n + s] + ed[g * n + d];
  a = a >= 0.f ? a : 0.2f * a;                    // leaky_relu 0.2
  alpha[idx] = a;
  atomicMaxF(&mbuf[g * n + d], a);
}

__global__ void gat_expsum_k(const int* __restrict__ ind, int Elist, int Etot, int n,
                             float* __restrict__ alpha, const float* __restrict__ mbuf,
                             float* __restrict__ sbuf)
{
  long idx = (long)blockIdx.x * blockDim.x + threadIdx.x;
  if (idx >= (long)G_ * Etot) return;
  long g = idx / Etot; int j = (int)(idx % Etot);
  int d = j < Elist ? ind[Elist + j] : j - Elist;
  float ex = expf(alpha[idx] - mbuf[g * n + d]);
  alpha[idx] = ex;
  atomicAdd(&sbuf[g * n + d], ex);
}

__global__ void gat_aggr_k(const int* __restrict__ ind, int Elist, int Etot, int n,
                           const float* __restrict__ alpha, const float* __restrict__ sbuf,
                           const float* __restrict__ Z, float* __restrict__ ACC)
{
  long idx = (long)blockIdx.x * blockDim.x + threadIdx.x;
  if (idx >= (long)G_ * Etot) return;
  long g = idx / Etot; int j = (int)(idx % Etot);
  int s = j < Elist ? ind[j] : j - Elist;
  int d = j < Elist ? ind[Elist + j] : j - Elist;
  float w = alpha[idx] / sbuf[g * n + d];
  const float* z = Z + (g * n + s) * (long)D_;
  float* o = ACC + (g * n + d) * (long)D_;
#pragma unroll 8
  for (int f = 0; f < D_; ++f) atomicAdd(&o[f], w * z[f]);
}

__global__ void gat_fin_k(const float* __restrict__ ACC, const float* __restrict__ bias,
                          float* __restrict__ X, int n)
{
  long idx = (long)blockIdx.x * blockDim.x + threadIdx.x;
  if (idx >= (long)G_ * n * D_) return;
  int f = (int)(idx % D_);
  float v = ACC[idx] + bias[f];
  X[idx] = v > 0.f ? v : 0.f;
}

// ---------------------------------------------------------------- temporal
// X[B,T,R,64] -> H[(rowOff + b*R + i)*1536 + d*24 + t]
__global__ void toT_k(const float* __restrict__ X, float* __restrict__ H,
                      int R, long rowOff)
{
  long idx = (long)blockIdx.x * blockDim.x + threadIdx.x;
  if (idx >= (long)G_ * R * D_) return;
  int d = (int)(idx % D_);
  long gi = idx / D_;
  long g = gi / R, i = gi % R;
  long b = g / T_; int t = (int)(g % T_);
  H[(rowOff + b * R + i) * (long)(D_ * T_) + (long)d * T_ + t] = X[idx];
}

__global__ void fromT_k(const float* __restrict__ H, float* __restrict__ X,
                        int R, long rowOff)
{
  long idx = (long)blockIdx.x * blockDim.x + threadIdx.x;
  if (idx >= (long)G_ * R * D_) return;
  int d = (int)(idx % D_);
  long gi = idx / D_;
  long g = gi / R, i = gi % R;
  long b = g / T_; int t = (int)(g % T_);
  X[idx] = H[(rowOff + b * R + i) * (long)(D_ * T_) + (long)d * T_ + t];
}

// causal dilated conv1d, K=3, OIH weights, relu
__global__ void conv_k(const float* __restrict__ in, const float* __restrict__ W,
                       const float* __restrict__ bias, float* __restrict__ out,
                       long Mrows, int dil)
{
  long idx = (long)blockIdx.x * blockDim.x + threadIdx.x;
  long total = Mrows * D_ * T_;
  if (idx >= total) return;
  int t = (int)(idx % T_);
  int o = (int)((idx / T_) % D_);
  long m = idx / (T_ * D_);
  const float* hrow = in + m * (long)(D_ * T_);
  const float* w = W + (long)o * (D_ * 3);
  float acc = bias[o];
  for (int c = 0; c < D_; ++c) {
#pragma unroll
    for (int k = 0; k < 3; ++k) {
      int tt = t - (2 - k) * dil;
      if (tt >= 0) acc += w[c * 3 + k] * hrow[c * T_ + tt];
    }
  }
  out[idx] = acc > 0.f ? acc : 0.f;
}

// ---------------------------------------------------------------- tail
__global__ void cumsum_res_k(const float* __restrict__ Y, const float* __restrict__ RES,
                             float* __restrict__ X, int R)
{
  long idx = (long)blockIdx.x * blockDim.x + threadIdx.x;
  if (idx >= (long)B_ * R * D_) return;
  int d = (int)(idx % D_);
  long bi = idx / D_;
  long b = bi / R, i = bi % R;
  float acc = 0.f;
  float r = RES[(b * R + i) * (long)D_ + d];
  for (int t = 0; t < T_; ++t) {
    long o = (((b * T_ + t) * R + i) * (long)D_) + d;
    acc += Y[o];
    float v = acc + r;
    X[o] = v > 0.f ? v : 0.f;
  }
}

__global__ void out_node_k(const float* __restrict__ X, const float* __restrict__ oW,
                           const float* __restrict__ ob, float* __restrict__ out)
{
  long idx = (long)blockIdx.x * blockDim.x + threadIdx.x;
  if (idx >= (long)G_ * N_) return;
  const float* r = X + idx * D_;
  float s = ob[0];
#pragma unroll 8
  for (int d = 0; d < D_; ++d) s += r[d] * oW[d];
  float v = s * (1.0f / 6.0f) + 0.5f;
  v = v < 0.f ? 0.f : (v > 1.f ? 1.f : v);
  out[idx] = v;
}

__global__ void out_edge_k(const float* __restrict__ Eb, const float* __restrict__ oW,
                           const float* __restrict__ ob, float* __restrict__ out)
{
  long idx = (long)blockIdx.x * blockDim.x + threadIdx.x;
  if (idx >= (long)G_ * E_) return;
  const float* r = Eb + idx * D_;
  float s = ob[0];
#pragma unroll 8
  for (int d = 0; d < D_; ++d) s += r[d] * oW[d];
  out[idx] = tanhf(s);
}

// ---------------------------------------------------------------- host
#define LA(kern, n, ...) kern<<<dim3((unsigned)(((n) + 255) / 256)), dim3(256), 0, stream>>>(__VA_ARGS__)

static inline void gemm(hipStream_t stream, const float* A, long sA, const float* Bt,
                        long sB, const float* bias, float* C, long sC, int ldC,
                        int transC, int rows, int K, int J, int relu)
{
  dim3 grid((rows + 16 * MT_ - 1) / (16 * MT_), J / 16, G_);
  gemm_bf16_wmma<<<grid, dim3(32), 0, stream>>>(A, sA, Bt, sB, bias, C, sC, ldC,
                                                transC, rows, K, J, relu);
}

extern "C" void kernel_launch(void* const* d_in, const int* in_sizes, int n_in,
                              void* d_out, int out_size, void* d_ws, size_t ws_size,
                              hipStream_t stream)
{
  (void)in_sizes; (void)n_in; (void)out_size; (void)ws_size;
  auto F = [&](int i) { return (const float*)d_in[i]; };
  const float *x_in = F(0), *e_in = F(1), *b_in = F(2), *a_in = F(3);
  const float *enc0W = F(4), *enc0b = F(5), *enc1W = F(6), *enc1b = F(7);
  const float *enc2W = F(8), *enc2b = F(9), *enc3W = F(10), *enc3b = F(11);
  const float *inci = F(12);
  const float *concW = F(49), *concb = F(50), *resW = F(51), *resb = F(52);
  const float *outW = F(53), *outb = F(54);
  const int *edge_ind = (const int*)d_in[55];
  const int *node_ind = (const int*)d_in[56];

  const float* sp[2][18];
  for (int i = 0; i < 18; ++i) { sp[0][i] = F(13 + i); sp[1][i] = F(31 + i); }

  // ---- workspace layout (floats)
  float* w = (float*)d_ws;
  auto take = [&](long n) { float* p = w; w += n; return p; };
  float* X    = take((long)G_ * N_ * D_);
  float* EBF  = take((long)G_ * E_ * D_);
  float* BBUF = take((long)G_ * N_ * 32);
  float* ABUF = take((long)G_ * E_ * 32);
  float* RESX = take((long)B_ * N_ * D_);
  float* RESE = take((long)B_ * E_ * D_);
  float* CATN = take((long)G_ * N_ * 96);
  float* CATE = take((long)G_ * E_ * 96);
  float* ZN   = take((long)G_ * N_ * D_);
  float* ZE   = take((long)G_ * E_ * D_);
  float* ACC  = take((long)G_ * E_ * D_);
  float* NE32 = take((long)G_ * E_ * 32);   // holds transposed [32, E|N] tiles
  float* WMSK = take((long)N_ * E_);
  float* ESB  = take((long)G_ * E_);
  float* EDB  = take((long)G_ * E_);
  float* ALPH = take((long)G_ * (M_ + E_));
  float* MBUF = take((long)G_ * E_);
  float* SBUF = take((long)G_ * E_);
  const long HROWS = (long)B_ * N_ + (long)B_ * E_;   // 7000
  float* H1 = take(HROWS * D_ * T_);
  float* H2 = take(HROWS * D_ * T_);
  // transposed weights [J, K] for the GEMM B operand
  float* WTgx[2][2]; float* WTge[2][2]; float* WTne[2][2]; float* WTen[2][2];
  for (int s = 0; s < 2; ++s)
    for (int l = 0; l < 2; ++l) {
      WTgx[s][l] = take(96 * 64);
      WTge[s][l] = take(96 * 64);
      WTne[s][l] = take(64 * 32);
      WTen[s][l] = take(64 * 32);
    }
  float* WTconc = take(96 * 64);
  float* WTres  = take(64 * 64);

  // ---- one-shot weight transposes
  for (int s = 0; s < 2; ++s)
    for (int l = 0; l < 2; ++l) {
      LA(transpose_k, (long)96 * 64, sp[s][0] + (long)l * 96 * 64, WTgx[s][l], 96, 64);
      LA(transpose_k, (long)96 * 64, sp[s][4] + (long)l * 96 * 64, WTge[s][l], 96, 64);
      LA(transpose_k, (long)64 * 32, sp[s][8] + (long)l * 64 * 32, WTne[s][l], 64, 32);
      LA(transpose_k, (long)64 * 32, sp[s][12] + (long)l * 64 * 32, WTen[s][l], 64, 32);
    }
  LA(transpose_k, (long)96 * 64, concW, WTconc, 96, 64);
  LA(transpose_k, (long)64 * 64, resW, WTres, 64, 64);

  // ---- GAT helper (segment softmax + aggregate), writes relu(out+bias)->Xout
  auto gat = [&](const float* Z, const float* as_, const float* ad_, const float* bias,
                 const int* ind, int Elist, int Etot, int n, float* Xout) {
    LA(fill_k, (long)G_ * n, MBUF, -3.0e38f, (long)G_ * n);
    LA(fill_k, (long)G_ * n, SBUF, 0.f, (long)G_ * n);
    LA(fill_k, (long)G_ * n * D_, ACC, 0.f, (long)G_ * n * D_);
    LA(gat_dots_k, (long)G_ * n, Z, as_, ad_, ESB, EDB, n);
    LA(gat_alpha_k, (long)G_ * Etot, ESB, EDB, ind, Elist, Etot, n, ALPH, MBUF);
    LA(gat_expsum_k, (long)G_ * Etot, ind, Elist, Etot, n, ALPH, MBUF, SBUF);
    LA(gat_aggr_k, (long)G_ * Etot, ind, Elist, Etot, n, ALPH, SBUF, Z, ACC);
    LA(gat_fin_k, (long)G_ * n * D_, ACC, bias, Xout, n);
  };

  // ---- spatio-temporal block
  auto stblock = [&](int s) {
    const float* const* p = sp[s];
    for (int l = 0; l < 2; ++l) {
      // xe = node_edge(e): t = relu(e@neW+neb) stored transposed [32,E],
      // then xe = (ne_w*|inc|+ne_p) @ t  -> CATN cols 64..95
      gemm(stream, EBF, (long)E_ * D_, WTne[s][l], 0, p[9] + l * 32,
           NE32, (long)32 * E_, E_, 1, E_, D_, 32, 1);
      LA(wmask_ne_k, (long)N_ * E_, p[10] + (long)l * N_ * E_, p[11] + (long)l * N_ * E_, inci, WMSK);
      gemm(stream, WMSK, 0, NE32, (long)32 * E_, nullptr,
           CATN + 64, (long)N_ * 96, 96, 0, N_, E_, 32, 0);
      // x-GAT on concat([x, xe])
      LA(pack_k, (long)G_ * N_ * D_, X, CATN, (long)G_ * N_, D_, 96, 0);
      gemm(stream, CATN, (long)N_ * 96, WTgx[s][l], 0, nullptr,
           ZN, (long)N_ * D_, D_, 0, N_, 96, D_, 0);
      gat(ZN, p[1] + l * D_, p[2] + l * D_, p[3] + l * D_, edge_ind, E_, E_ + N_, N_, X);
      // ex = node_edge(x) with inc^T: t = relu(x@enW+enb) transposed [32,N]
      gemm(stream, X, (long)N_ * D_, WTen[s][l], 0, p[13] + l * 32,
           NE32, (long)32 * N_, N_, 1, N_, D_, 32, 1);
      LA(wmask_en_k, (long)E_ * N_, p[14] + (long)l * E_ * N_, p[15] + (long)l * E_ * N_, inci, WMSK);
      gemm(stream, WMSK, 0, NE32, (long)32 * N_, nullptr,
           CATE + 64, (long)E_ * 96, 96, 0, E_, N_, 32, 0);
      // e-GAT on concat([e, ex])
      LA(pack_k, (long)G_ * E_ * D_, EBF, CATE, (long)G_ * E_, D_, 96, 0);
      gemm(stream, CATE, (long)E_ * 96, WTge[s][l], 0, nullptr,
           ZE, (long)E_ * D_, D_, 0, E_, 96, D_, 0);
      gat(ZE, p[5] + l * D_, p[6] + l * D_, p[7] + l * D_, node_ind, M_, M_ + E_, E_, EBF);
    }
    // temporal: [B,T,R,D] -> [(b r), D, T], two causal dilated convs, back
    LA(toT_k, (long)G_ * N_ * D_, X, H1, N_, 0L);
    LA(toT_k, (long)G_ * E_ * D_, EBF, H1, E_, (long)B_ * N_);
    LA(conv_k, HROWS * D_ * T_, H1, p[16], p[17], H2, HROWS, 1);
    LA(conv_k, HROWS * D_ * T_, H2, p[16] + D_ * D_ * 3, p[17] + D_, H1, HROWS, 2);
    LA(fromT_k, (long)G_ * N_ * D_, H1, X, N_, 0L);
    LA(fromT_k, (long)G_ * E_ * D_, H1, EBF, E_, (long)B_ * N_);
  };

  // ---- encoders (capture pre-relu last-timestep residuals)
  LA(encode_k, (long)G_ * N_ * D_, x_in, enc0W, enc0b, X, RESX, N_, 4, D_);
  LA(encode_k, (long)G_ * E_ * D_, e_in, enc1W, enc1b, EBF, RESE, E_, 3, D_);
  LA(encode_k, (long)G_ * N_ * 32, b_in, enc2W, enc2b, BBUF, (float*)nullptr, N_, 4, 32);
  LA(encode_k, (long)G_ * E_ * 32, a_in, enc3W, enc3b, ABUF, (float*)nullptr, E_, 3, 32);

  stblock(0);

  // ---- conc: relu(concat([x, bb]) @ conc_W + conc_b), shared weights
  LA(pack_k, (long)G_ * N_ * D_, X, CATN, (long)G_ * N_, D_, 96, 0);
  LA(pack_k, (long)G_ * N_ * 32, BBUF, CATN, (long)G_ * N_, 32, 96, 64);
  gemm(stream, CATN, (long)N_ * 96, WTconc, 0, concb, X, (long)N_ * D_, D_, 0, N_, 96, D_, 1);
  LA(pack_k, (long)G_ * E_ * D_, EBF, CATE, (long)G_ * E_, D_, 96, 0);
  LA(pack_k, (long)G_ * E_ * 32, ABUF, CATE, (long)G_ * E_, 32, 96, 64);
  gemm(stream, CATE, (long)E_ * 96, WTconc, 0, concb, EBF, (long)E_ * D_, D_, 0, E_, 96, D_, 1);

  stblock(1);

  // ---- residual head: relu(cumsum_T(x @ res_W + res_b) + res)
  gemm(stream, X, (long)N_ * D_, WTres, 0, resb, ZN, (long)N_ * D_, D_, 0, N_, D_, D_, 0);
  LA(cumsum_res_k, (long)B_ * N_ * D_, ZN, RESX, X, N_);
  gemm(stream, EBF, (long)E_ * D_, WTres, 0, resb, ZE, (long)E_ * D_, D_, 0, E_, D_, D_, 0);
  LA(cumsum_res_k, (long)B_ * E_ * D_, ZE, RESE, EBF, E_);

  // ---- outputs: hardsigmoid(nodes), tanh(edges), concatenated flat
  float* out = (float*)d_out;
  LA(out_node_k, (long)G_ * N_, X, outW, outb, out);
  LA(out_edge_k, (long)G_ * E_, EBF, outW, outb, out + (long)G_ * N_);
}
